// MambaBlock_59828894433827
// MI455X (gfx1250) — matrix operand order repaired and checked by reading, not compiled, not run on hardware
//
#include <hip/hip_runtime.h>

// ---------- types ----------
typedef __attribute__((ext_vector_type(16))) __bf16 v16bf;
typedef __attribute__((ext_vector_type(8)))  __bf16 bf16x8;
typedef __attribute__((ext_vector_type(8)))  float  v8f;
typedef int v4i_ __attribute__((vector_size(16)));

static __device__ __forceinline__ __bf16 to_bf16(float f) {
  unsigned u = __builtin_bit_cast(unsigned, f);
  unsigned r = u + 0x7FFFu + ((u >> 16) & 1u);   // round-to-nearest-even
  unsigned short h = (unsigned short)(r >> 16);
  return __builtin_bit_cast(__bf16, h);
}

// ---------- CDNA5 async global->LDS staging (ASYNCcnt path) ----------
#if __has_builtin(__builtin_amdgcn_global_load_async_to_lds_b128)
#define USE_ASYNC_LDS 1
#else
#define USE_ASYNC_LDS 0
#endif

#if __has_builtin(__builtin_amdgcn_s_wait_asynccnt)
#define WAIT_ASYNCCNT0() __builtin_amdgcn_s_wait_asynccnt(0)
#else
#define WAIT_ASYNCCNT0() asm volatile("s_wait_asynccnt 0" ::: "memory")
#endif

// copy one aligned 16-byte chunk global -> LDS
static __device__ __forceinline__ void stage16(const __bf16* __restrict__ g,
                                               __bf16* l) {
#if USE_ASYNC_LDS
  __builtin_amdgcn_global_load_async_to_lds_b128(
      (__attribute__((address_space(1))) v4i_*)(v4i_*)(void*)g,
      (__attribute__((address_space(3))) v4i_*)(v4i_*)(void*)l,
      0, 0);
#else
  *(bf16x8*)l = *(const bf16x8*)g;
#endif
}

// ---------- fp32 -> bf16 conversion ----------
__global__ __launch_bounds__(256) void cvt_bf16_kernel(
    const float* __restrict__ in, __bf16* __restrict__ out, int n) {
  int i = blockIdx.x * 256 + threadIdx.x;
  if (i < n) out[i] = to_bf16(in[i]);
}

// ---------- LayerNorm (one block per token) ----------
__global__ __launch_bounds__(256) void layernorm_kernel(
    const float* __restrict__ in, const float* __restrict__ g,
    const float* __restrict__ b, float* __restrict__ outF,
    __bf16* __restrict__ outB, int Hdim) {
  const int row = blockIdx.x;
  const float* xr = in + (size_t)row * Hdim;
  float s = 0.f, ss = 0.f;
  for (int i = threadIdx.x; i < Hdim; i += 256) {
    float v = xr[i];
    s += v; ss += v * v;
  }
  for (int o = 16; o > 0; o >>= 1) {
    s  += __shfl_xor(s,  o, 32);
    ss += __shfl_xor(ss, o, 32);
  }
  __shared__ float sh_s[8], sh_ss[8];
  __shared__ float sh_mu, sh_rs;
  int wid = threadIdx.x >> 5, lane = threadIdx.x & 31;
  if (lane == 0) { sh_s[wid] = s; sh_ss[wid] = ss; }
  __syncthreads();
  if (threadIdx.x == 0) {
    float S = 0.f, SS = 0.f;
    for (int i = 0; i < 8; ++i) { S += sh_s[i]; SS += sh_ss[i]; }
    float mu  = S / (float)Hdim;
    float var = SS / (float)Hdim - mu * mu;
    sh_mu = mu;
    sh_rs = rsqrtf(var + 1e-5f);
  }
  __syncthreads();
  float mu = sh_mu, rs = sh_rs;
  for (int i = threadIdx.x; i < Hdim; i += 256) {
    float v = (xr[i] - mu) * rs * g[i] + b[i];
    size_t idx = (size_t)row * Hdim + i;
    if (outF) outF[idx] = v;
    if (outB) outB[idx] = to_bf16(v);
  }
}

// ---------- WMMA GEMM: out[T,O] = epi(A[T,K] @ W[O,K]^T + bias) ----------
// A, W bf16 row-major. BM=128 fixed. 256 threads = 8 waves.
// BN = block tile N, WN = wave tile N.
// EPI: 0 = +bias -> f32; 1 = softplus(+bias) -> f32;
//      2 = gelu(+bias) -> bf16; 3 = +bias +extra -> f32 (residual add)
template <int BN, int WN, int EPI>
__global__ __launch_bounds__(256) void gemm_wmma_kernel(
    const __bf16* __restrict__ A, const __bf16* __restrict__ W,
    const float* __restrict__ bias, const float* __restrict__ extra,
    float* __restrict__ outF, __bf16* __restrict__ outB,
    int K, int O) {
  constexpr int BM    = 128;
  constexpr int WCOLS = BN / WN;        // waves along N
  constexpr int WROWS = 8 / WCOLS;      // waves along M
  constexpr int WTM   = BM / WROWS;     // rows per wave tile
  constexpr int AF    = WTM / 16;       // A-fragments per wave
  constexpr int BF    = WN / 16;        // B-fragments per wave
  static_assert(WCOLS * WROWS == 8, "bad wave grid");

  __shared__ alignas(16) __bf16 At[BM][40];  // 40-elem stride: 80B, 16B aligned
  __shared__ alignas(16) __bf16 Bt[BN][40];

  const int tid    = threadIdx.x;
  const int wid    = tid >> 5;
  const int lane   = tid & 31;
  const int laneHi = lane >> 4;          // 0 or 1
  const int lane16 = lane & 15;
  const int wrow   = wid % WROWS;
  const int wcol   = wid / WROWS;

  const int rowBase = blockIdx.x * BM;
  const int colBase = blockIdx.y * BN;

  v8f acc[AF][BF];
  for (int i = 0; i < AF; ++i)
    for (int j = 0; j < BF; ++j)
      for (int r = 0; r < 8; ++r) acc[i][j][r] = 0.f;

  union F16 { v16bf v; bf16x8 h[2]; };

  for (int k0 = 0; k0 < K; k0 += 32) {
    __syncthreads();
    // cooperative staging: 16 bf16 (2 x 16B) per unit
    for (int u = tid; u < BM * 2; u += 256) {
      int r = u >> 1, koff = (u & 1) * 16;
      const __bf16* src = A + (size_t)(rowBase + r) * K + k0 + koff;
      stage16(src,     &At[r][koff]);
      stage16(src + 8, &At[r][koff + 8]);
    }
    for (int u = tid; u < BN * 2; u += 256) {
      int r = u >> 1, koff = (u & 1) * 16;
      const __bf16* src = W + (size_t)(colBase + r) * K + k0 + koff;
      stage16(src,     &Bt[r][koff]);
      stage16(src + 8, &Bt[r][koff + 8]);
    }
#if USE_ASYNC_LDS
    WAIT_ASYNCCNT0();
#endif
    __syncthreads();

    // B fragments: lane -> column (lane&15), 16 consecutive K at 16*(lane>>4)
    F16 bfr[BF];
    for (int j = 0; j < BF; ++j) {
      int nn = wcol * WN + j * 16 + lane16;
      int kb = laneHi * 16;
      bfr[j].h[0] = *(const bf16x8*)&Bt[nn][kb];
      bfr[j].h[1] = *(const bf16x8*)&Bt[nn][kb + 8];
    }
    // A fragments: lane -> row (lane&15), 16 consecutive K at 16*(lane>>4)
    for (int i = 0; i < AF; ++i) {
      int rr = wrow * WTM + i * 16 + lane16;
      int kg = laneHi * 16;
      F16 afr;
      afr.h[0] = *(const bf16x8*)&At[rr][kg];
      afr.h[1] = *(const bf16x8*)&At[rr][kg + 8];
      for (int j = 0; j < BF; ++j) {
        acc[i][j] = __builtin_amdgcn_wmma_f32_16x16x32_bf16(
            false, afr.v, false, bfr[j].v, (short)0, acc[i][j], false, false);
      }
    }
  }

  // epilogue: D tile element (m = r + 8*laneHi, n = lane&15)
  for (int i = 0; i < AF; ++i) {
    for (int j = 0; j < BF; ++j) {
      int col = colBase + wcol * WN + j * 16 + lane16;
      float bv = bias[col];
      for (int r = 0; r < 8; ++r) {
        int row = rowBase + wrow * WTM + i * 16 + r + laneHi * 8;
        size_t idx = (size_t)row * O + col;
        float v = acc[i][j][r] + bv;
        if (EPI == 1) {
          v = (v > 20.f) ? v : log1pf(__expf(v));
          outF[idx] = v;
        } else if (EPI == 2) {
          float gv = 0.5f * v * (1.f + erff(v * 0.70710678118f));
          outB[idx] = to_bf16(gv);
        } else if (EPI == 3) {
          outF[idx] = v + extra[idx];
        } else {
          outF[idx] = v;
        }
      }
    }
  }
}

// ---------- selective SSM scan ----------
// one lane per (b, h, n); 16-lane shfl-xor tree reduces over N.
// Software-pipelined: loads of step t+1 overlap the reduction of step t.
// ssm_out = x + y + D*xn
__global__ __launch_bounds__(256) void ssm_scan_kernel(
    const float* __restrict__ x, const float* __restrict__ xn,
    const float* __restrict__ delta, const float* __restrict__ bc,
    const float* __restrict__ A_log, const float* __restrict__ D,
    float* __restrict__ ssm_out) {
  const int tid = blockIdx.x * 256 + threadIdx.x;   // 0 .. 32767
  const int n = tid & 15;
  const int h = (tid >> 4) & 1023;
  const int b = tid >> 14;
  const float A_hn = -__expf(A_log[h * 16 + n]);
  const float D_h  = D[h];
  float state = 0.f;
  const size_t baseTok = (size_t)b * 2048;

  size_t tk = baseTok;
  float dlt = delta[tk * 1024 + h];
  float xv  = xn[tk * 1024 + h];
  float Bn  = bc[tk * 32 + n];
  float Cn  = bc[tk * 32 + 16 + n];

  for (int t = 0; t < 2048; ++t) {
    // prefetch next step (independent of the recurrence chain)
    float dlt_n = 0.f, xv_n = 0.f, Bn_n = 0.f, Cn_n = 0.f;
    if (t + 1 < 2048) {
      size_t tk1 = baseTok + t + 1;
      dlt_n = delta[tk1 * 1024 + h];
      xv_n  = xn[tk1 * 1024 + h];
      Bn_n  = bc[tk1 * 32 + n];
      Cn_n  = bc[tk1 * 32 + 16 + n];
    }
    float dA = __expf(dlt * A_hn);
    state = fmaf(dA, state, dlt * Bn * xv);
    float yc = state * Cn;
    yc += __shfl_xor(yc, 1, 16);
    yc += __shfl_xor(yc, 2, 16);
    yc += __shfl_xor(yc, 4, 16);
    yc += __shfl_xor(yc, 8, 16);
    if (n == 0) {
      size_t idx = (baseTok + t) * 1024 + h;
      ssm_out[idx] = x[idx] + yc + D_h * xv;
    }
    dlt = dlt_n; xv = xv_n; Bn = Bn_n; Cn = Cn_n;
  }
}

// ---------- host launch ----------
extern "C" void kernel_launch(void* const* d_in, const int* in_sizes, int n_in,
                              void* d_out, int out_size, void* d_ws, size_t ws_size,
                              hipStream_t stream) {
  (void)in_sizes; (void)n_in; (void)out_size; (void)ws_size;
  constexpr int H = 1024, N = 16, F = 4096;
  constexpr int T = 2 * 2048;                 // B*L tokens

  const float* x     = (const float*)d_in[0];
  const float* ln1_g = (const float*)d_in[1];
  const float* ln1_b = (const float*)d_in[2];
  const float* Wd    = (const float*)d_in[3];
  const float* bd    = (const float*)d_in[4];
  const float* Wbc   = (const float*)d_in[5];
  const float* bbc   = (const float*)d_in[6];
  const float* A_log = (const float*)d_in[7];
  const float* Dv    = (const float*)d_in[8];
  const float* ln2_g = (const float*)d_in[9];
  const float* ln2_b = (const float*)d_in[10];
  const float* W1    = (const float*)d_in[11];
  const float* b1    = (const float*)d_in[12];
  const float* W2    = (const float*)d_in[13];
  const float* b2    = (const float*)d_in[14];
  float* out = (float*)d_out;

  // workspace layout (all sizes multiples of 256B)
  char* ws = (char*)d_ws;
  size_t off = 0;
  auto alloc = [&](size_t bytes) { char* p = ws + off; off += bytes; return p; };
  float*  xn_f32   = (float*) alloc((size_t)T * H * 4);   // 16 MB
  __bf16* xn_bf    = (__bf16*)alloc((size_t)T * H * 2);   // 8 MB
  __bf16* Wd_bf    = (__bf16*)alloc((size_t)H * H * 2);   // 2 MB
  __bf16* Wbc_bf   = (__bf16*)alloc((size_t)2 * N * H * 2);
  __bf16* W1_bf    = (__bf16*)alloc((size_t)F * H * 2);   // 8 MB
  __bf16* W2_bf    = (__bf16*)alloc((size_t)H * F * 2);   // 8 MB
  float*  delta    = (float*) alloc((size_t)T * H * 4);   // 16 MB
  float*  bcbuf    = (float*) alloc((size_t)T * 2 * N * 4);
  float*  ssm_out  = (float*) alloc((size_t)T * H * 4);   // 16 MB
  __bf16* norm_bf  = (__bf16*)alloc((size_t)T * H * 2);   // 8 MB
  __bf16* hff_bf   = (__bf16*)alloc((size_t)T * F * 2);   // 32 MB

  // 1) weight conversions fp32 -> bf16
  {
    int n;
    n = H * H;     cvt_bf16_kernel<<<(n + 255) / 256, 256, 0, stream>>>(Wd,  Wd_bf,  n);
    n = 2 * N * H; cvt_bf16_kernel<<<(n + 255) / 256, 256, 0, stream>>>(Wbc, Wbc_bf, n);
    n = F * H;     cvt_bf16_kernel<<<(n + 255) / 256, 256, 0, stream>>>(W1,  W1_bf,  n);
    n = H * F;     cvt_bf16_kernel<<<(n + 255) / 256, 256, 0, stream>>>(W2,  W2_bf,  n);
  }

  // 2) LayerNorm 1 -> xn (f32 + bf16)
  layernorm_kernel<<<T, 256, 0, stream>>>(x, ln1_g, ln1_b, xn_f32, xn_bf, H);

  // 3) delta = softplus(xn @ Wd^T + bd)      [T x H]
  gemm_wmma_kernel<256, 64, 1><<<dim3(T / 128, H / 256), 256, 0, stream>>>(
      xn_bf, Wd_bf, bd, nullptr, delta, nullptr, H, H);

  // 4) bc = xn @ Wbc^T + bbc                  [T x 32]
  gemm_wmma_kernel<32, 32, 0><<<dim3(T / 128, 1), 256, 0, stream>>>(
      xn_bf, Wbc_bf, bbc, nullptr, bcbuf, nullptr, H, 2 * N);

  // 5) selective scan -> ssm_out = x + y + D*xn
  ssm_scan_kernel<<<(2 * H * N) / 256, 256, 0, stream>>>(
      x, xn_f32, delta, bcbuf, A_log, Dv, ssm_out);

  // 6) LayerNorm 2 -> normed (bf16 only)
  layernorm_kernel<<<T, 256, 0, stream>>>(ssm_out, ln2_g, ln2_b, nullptr, norm_bf, H);

  // 7) hff = gelu(normed @ W1^T + b1)         [T x F], bf16
  gemm_wmma_kernel<256, 64, 2><<<dim3(T / 128, F / 256), 256, 0, stream>>>(
      norm_bf, W1_bf, b1, nullptr, nullptr, hff_bf, H, F);

  // 8) out = hff @ W2^T + b2 + ssm_out        [T x H], f32
  gemm_wmma_kernel<256, 64, 3><<<dim3(T / 128, H / 256), 256, 0, stream>>>(
      hff_bf, W2_bf, b2, ssm_out, out, nullptr, F, H);
}
